// XMoETransformerLayer_52948356825288
// MI455X (gfx1250) — matrix operand order repaired and checked by reading, not compiled, not run on hardware
//
#include <hip/hip_runtime.h>
#include <hip/hip_bf16.h>
#include <math.h>

// ---------------------------------------------------------------------------
// XMoE transformer layer for MI455X (gfx1250, wave32, WMMA).
// Heavy GEMMs: V_WMMA_F32_16X16X32_BF16, fp32 accumulate, with double-buffered
// GLOBAL_LOAD_ASYNC_TO_LDS_B128 staging (ASYNCcnt pipeline).
// ---------------------------------------------------------------------------

typedef __attribute__((ext_vector_type(16))) __bf16 v16bf;
typedef __attribute__((ext_vector_type(8)))  float  v8f;

#define S_LEN   2048
#define B_SZ    2
#define D_MODEL 1024
#define N_HEAD  16
#define D_HEAD  64
#define DFF_SZ  4096
#define N_EXP   4
#define T_TOK   (S_LEN * B_SZ)   // 4096

constexpr int EPI_F32       = 0;  // Cf = alpha*acc
constexpr int EPI_BF16      = 1;  // Cb = bf16(alpha*acc)
constexpr int EPI_GELU_BF16 = 2;  // Cb = bf16(gelu(acc + bias[n]))
constexpr int EPI_ATOMIC    = 3;  // atomicAdd(Cf, rowscale[m]*(acc + bias[n]))

// ---- CDNA5 async global->LDS copy (16B per lane), tracked by ASYNCcnt ----
__device__ __forceinline__ void async_ld_b128(unsigned lds_addr, const void* gptr)
{
    asm volatile("global_load_async_to_lds_b128 %0, %1, off"
                 :: "v"(lds_addr), "v"(gptr) : "memory");
}
__device__ __forceinline__ void wait_async_le4()
{
    asm volatile("s_wait_asynccnt 0x4" ::: "memory");
}
__device__ __forceinline__ void wait_async_0()
{
    asm volatile("s_wait_asynccnt 0x0" ::: "memory");
}
__device__ __forceinline__ unsigned lds_off(const void* p)
{
    return (unsigned)(size_t)p;   // low 32 bits of flat LDS address = LDS offset
}

// ---------------------------------------------------------------------------
// Tiled WMMA GEMM: C[M,N] = A[M,K] * B^T, A is [M][K] (lda), B is [N][K] (ldb).
// Block = 256 threads (8 waves), tile 128x128, K-step 32, LDS double-buffered
// via async copies. NO bounds checks on staging: caller guarantees all
// 128x32 tiles are readable (buffers padded where needed).
// ---------------------------------------------------------------------------
template<int EPI>
__global__ __launch_bounds__(256)
void gemm_wmma(int M, int N, int K,
               const __bf16* __restrict__ A, int lda,
               const __bf16* __restrict__ B, int ldb,
               float* __restrict__ Cf, __bf16* __restrict__ Cb, int ldc,
               float alpha,
               const float* __restrict__ bias,
               const float* __restrict__ rowscale, int rs_stride)
{
    constexpr int BM = 128, BN = 128, BK = 32, LSTR = 40; // 40 keeps 16B align
    __shared__ __bf16 As[2][BM][LSTR];
    __shared__ __bf16 Bs[2][BN][LSTR];

    const int tid    = threadIdx.x;
    const int wave   = tid >> 5;
    const int lane   = tid & 31;
    const int laneLo = lane & 15;
    const int laneHi = lane >> 4;
    const int wm     = wave >> 2;   // 0..1
    const int wn     = wave & 3;    // 0..3
    const int m0     = blockIdx.y * BM;
    const int n0     = blockIdx.x * BN;

    // staging coordinates: each thread owns rows (rr, rr+64) at column cc
    const int rr = tid >> 2;          // 0..63
    const int cc = (tid & 3) << 3;    // 0,8,16,24

    union Frag { uint4 q[2]; v16bf v; };

    v8f acc[4][2];
#pragma unroll
    for (int a = 0; a < 4; a++)
#pragma unroll
        for (int b2 = 0; b2 < 2; b2++)
            acc[a][b2] = (v8f){0.f,0.f,0.f,0.f,0.f,0.f,0.f,0.f};

    // Issue one tile batch (4 async b128 instructions per wave).
    auto stage = [&](int p, int k0) {
        const __bf16* ga = A + (size_t)(m0 + rr) * lda + k0 + cc;
        async_ld_b128(lds_off(&As[p][rr][cc]),      ga);
        async_ld_b128(lds_off(&As[p][rr + 64][cc]), ga + (size_t)64 * lda);
        const __bf16* gb = B + (size_t)(n0 + rr) * ldb + k0 + cc;
        async_ld_b128(lds_off(&Bs[p][rr][cc]),      gb);
        async_ld_b128(lds_off(&Bs[p][rr + 64][cc]), gb + (size_t)64 * ldb);
    };

    const int nk = K / BK;
    stage(0, 0);

    for (int i = 0; i < nk; i++) {
        const int p = i & 1;
        if (i + 1 < nk) {            // prefetch next tile into other buffer
            stage(p ^ 1, (i + 1) * BK);
            wait_async_le4();        // in-order completion: batch i is done
        } else {
            wait_async_0();
        }
        __syncthreads();             // batch i visible to all waves

        Frag fa[4], fb[2];
#pragma unroll
        for (int a = 0; a < 4; a++) {
            int row = wm * 64 + a * 16 + laneLo;
            int kb  = laneHi * 8;               // lanes16-31 start at K=8
            fa[a].q[0] = *reinterpret_cast<const uint4*>(&As[p][row][kb]);
            fa[a].q[1] = *reinterpret_cast<const uint4*>(&As[p][row][kb + 16]);
        }
#pragma unroll
        for (int b2 = 0; b2 < 2; b2++) {
            int col = wn * 32 + b2 * 16 + laneLo;
            int kb  = laneHi * 8;               // lanes16-31 start at K=8
            fb[b2].q[0] = *reinterpret_cast<const uint4*>(&Bs[p][col][kb]);
            fb[b2].q[1] = *reinterpret_cast<const uint4*>(&Bs[p][col][kb + 16]);
        }
#pragma unroll
        for (int a = 0; a < 4; a++)
#pragma unroll
            for (int b2 = 0; b2 < 2; b2++)
                acc[a][b2] = __builtin_amdgcn_wmma_f32_16x16x32_bf16(
                    false, fa[a].v, false, fb[b2].v,
                    (short)0, acc[a][b2], false, false);

        __syncthreads();             // all reads of buffer p done -> reusable
    }

    // ---- epilogue: C/D layout = VGPR r -> (M = r + 8*laneHi, N = laneLo) ----
#pragma unroll
    for (int a = 0; a < 4; a++) {
        int mBase = m0 + wm * 64 + a * 16 + laneHi * 8;
#pragma unroll
        for (int b2 = 0; b2 < 2; b2++) {
            int gn = n0 + wn * 32 + b2 * 16 + laneLo;
#pragma unroll
            for (int r = 0; r < 8; r++) {
                int gm = mBase + r;
                if (gm < M && gn < N) {
                    float v = acc[a][b2][r];
                    if constexpr (EPI == EPI_F32) {
                        Cf[(size_t)gm * ldc + gn] = alpha * v;
                    } else if constexpr (EPI == EPI_BF16) {
                        Cb[(size_t)gm * ldc + gn] = (__bf16)(alpha * v);
                    } else if constexpr (EPI == EPI_GELU_BF16) {
                        float x = v + bias[gn];
                        float g = 0.5f * x * (1.0f + erff(x * 0.7071067811865475f));
                        Cb[(size_t)gm * ldc + gn] = (__bf16)g;
                    } else { // EPI_ATOMIC
                        float x = (v + bias[gn]) * rowscale[(size_t)gm * rs_stride];
                        atomicAdd(&Cf[(size_t)gm * ldc + gn], x);
                    }
                }
            }
        }
    }
}

// ---------------------------------------------------------------------------
// Weight cast + transpose: in fp32 [R][C] -> out bf16 [C][R]. 32x32 LDS tiles.
// blockIdx.z selects expert slice (strides in elements).
// ---------------------------------------------------------------------------
__global__ __launch_bounds__(256)
void cast_transpose_kernel(const float* __restrict__ in, __bf16* __restrict__ out,
                           int R, int C, size_t inEStride, size_t outEStride)
{
    __shared__ __bf16 tile[32][33];
    const float* ip = in  + (size_t)blockIdx.z * inEStride;
    __bf16*      op = out + (size_t)blockIdx.z * outEStride;
    int r0 = blockIdx.y * 32, c0 = blockIdx.x * 32;
#pragma unroll
    for (int i = 0; i < 4; i++) {
        int r = threadIdx.y * 4 + i;
        tile[r][threadIdx.x] = (__bf16)ip[(size_t)(r0 + r) * C + c0 + threadIdx.x];
    }
    __syncthreads();
#pragma unroll
    for (int i = 0; i < 4; i++) {
        int c = threadIdx.y * 4 + i;
        op[(size_t)(c0 + c) * R + r0 + threadIdx.x] = tile[threadIdx.x][c];
    }
}

// V slice transpose: kv[(i,b,h), 2*DH] v-part [S][DH] (ld 4096) -> vT per (b,h)
// [DH][S] (ld 2048), padded to 128 rows per (b,h).
__global__ __launch_bounds__(256)
void transpose_v_kernel(const __bf16* __restrict__ kv, __bf16* __restrict__ vT)
{
    __shared__ __bf16 tile[32][33];
    const int bh = blockIdx.z;                 // b*16+h
    const int b = bh >> 4, h = bh & 15;
    const __bf16* in = kv + (size_t)b * (2 * D_MODEL) + h * (2 * D_HEAD) + D_HEAD;
    __bf16* out = vT + (size_t)bh * 128 * S_LEN;
    int r0 = blockIdx.y * 32, c0 = blockIdx.x * 32;
#pragma unroll
    for (int i = 0; i < 4; i++) {
        int r = threadIdx.y * 4 + i;
        tile[r][threadIdx.x] = in[(size_t)(r0 + r) * (B_SZ * 2 * D_MODEL) + c0 + threadIdx.x];
    }
    __syncthreads();
#pragma unroll
    for (int i = 0; i < 4; i++) {
        int c = threadIdx.y * 4 + i;
        out[(size_t)(c0 + c) * S_LEN + r0 + threadIdx.x] = tile[threadIdx.x][c];
    }
}

// ---------------------------------------------------------------------------
// LayerNorm over D=1024 with optional residual / pre-LN sum / bf16 output.
// ---------------------------------------------------------------------------
__global__ __launch_bounds__(256)
void layernorm_kernel(const float* __restrict__ in1, const float* __restrict__ in2,
                      const float* __restrict__ g, const float* __restrict__ b,
                      float eps, float* __restrict__ outSum,
                      float* __restrict__ outF, __bf16* __restrict__ outB)
{
    __shared__ float red[256];
    const size_t base = (size_t)blockIdx.x * D_MODEL;
    float x[4];
    float s = 0.f;
#pragma unroll
    for (int i = 0; i < 4; i++) {
        int c = threadIdx.x + i * 256;
        float v = in1[base + c];
        if (in2) v += in2[base + c];
        x[i] = v; s += v;
    }
    red[threadIdx.x] = s; __syncthreads();
    for (int o = 128; o > 0; o >>= 1) {
        if (threadIdx.x < o) red[threadIdx.x] += red[threadIdx.x + o];
        __syncthreads();
    }
    float mean = red[0] * (1.0f / D_MODEL);
    __syncthreads();
    float sq = 0.f;
#pragma unroll
    for (int i = 0; i < 4; i++) { float d = x[i] - mean; sq += d * d; }
    red[threadIdx.x] = sq; __syncthreads();
    for (int o = 128; o > 0; o >>= 1) {
        if (threadIdx.x < o) red[threadIdx.x] += red[threadIdx.x + o];
        __syncthreads();
    }
    float rstd = rsqrtf(red[0] * (1.0f / D_MODEL) + eps);
#pragma unroll
    for (int i = 0; i < 4; i++) {
        int c = threadIdx.x + i * 256;
        if (outSum) outSum[base + c] = x[i];
        float y = (x[i] - mean) * rstd * g[c] + b[c];
        if (outF) outF[base + c] = y;
        if (outB) outB[base + c] = (__bf16)y;
    }
}

// Row softmax over N=2048 (attention axis=1), bf16 output.
__global__ __launch_bounds__(256)
void softmax_row_kernel(const float* __restrict__ in, __bf16* __restrict__ out)
{
    __shared__ float red[256];
    const size_t base = (size_t)blockIdx.x * S_LEN;
    float v[8];
    float m = -3.4e38f;
#pragma unroll
    for (int i = 0; i < 8; i++) {
        v[i] = in[base + threadIdx.x + i * 256];
        m = fmaxf(m, v[i]);
    }
    red[threadIdx.x] = m; __syncthreads();
    for (int o = 128; o > 0; o >>= 1) {
        if (threadIdx.x < o) red[threadIdx.x] = fmaxf(red[threadIdx.x], red[threadIdx.x + o]);
        __syncthreads();
    }
    m = red[0]; __syncthreads();
    float s = 0.f;
#pragma unroll
    for (int i = 0; i < 8; i++) { v[i] = __expf(v[i] - m); s += v[i]; }
    red[threadIdx.x] = s; __syncthreads();
    for (int o = 128; o > 0; o >>= 1) {
        if (threadIdx.x < o) red[threadIdx.x] += red[threadIdx.x + o];
        __syncthreads();
    }
    float inv = 1.f / red[0];
#pragma unroll
    for (int i = 0; i < 8; i++)
        out[base + threadIdx.x + i * 256] = (__bf16)(v[i] * inv);
}

// MoE gating: probs=softmax(x@Wg), top-2 combine weights, aux-loss stats.
__global__ __launch_bounds__(256)
void gate_kernel(const float* __restrict__ x, const float* __restrict__ Wg,
                 float* __restrict__ cw, float* __restrict__ stats)
{
    int t = blockIdx.x * 256 + threadIdx.x;
    if (t >= T_TOK) return;
    const float* xr = x + (size_t)t * D_MODEL;
    float a0 = 0.f, a1 = 0.f, a2 = 0.f, a3 = 0.f;
    for (int d = 0; d < D_MODEL; d++) {
        float xv = xr[d];
        float4 w = *reinterpret_cast<const float4*>(Wg + (size_t)d * N_EXP);
        a0 += xv * w.x; a1 += xv * w.y; a2 += xv * w.z; a3 += xv * w.w;
    }
    float acc[4] = {a0, a1, a2, a3};
    float m = fmaxf(fmaxf(acc[0], acc[1]), fmaxf(acc[2], acc[3]));
    float p[4], s = 0.f;
#pragma unroll
    for (int e = 0; e < 4; e++) { p[e] = __expf(acc[e] - m); s += p[e]; }
#pragma unroll
    for (int e = 0; e < 4; e++) p[e] /= s;
    int i1 = 0;
    for (int e = 1; e < 4; e++) if (p[e] > p[i1]) i1 = e;
    int i2 = -1;
    for (int e = 0; e < 4; e++) if (e != i1 && (i2 < 0 || p[e] > p[i2])) i2 = e;
    float denom = p[i1] + p[i2];
#pragma unroll
    for (int e = 0; e < 4; e++) cw[(size_t)t * N_EXP + e] = 0.f;
    cw[(size_t)t * N_EXP + i1] = p[i1] / denom;
    cw[(size_t)t * N_EXP + i2] = p[i2] / denom;
#pragma unroll
    for (int e = 0; e < 4; e++) atomicAdd(&stats[e], p[e]);
    atomicAdd(&stats[4 + i1], 1.0f);
}

__global__ void loss_kernel(const float* __restrict__ stats, float* __restrict__ out)
{
    if (blockIdx.x == 0 && threadIdx.x == 0) {
        float l = 0.f;
        for (int e = 0; e < N_EXP; e++)
            l += (stats[e] / T_TOK) * (stats[4 + e] / T_TOK);
        *out = (float)N_EXP * l;
    }
}

__global__ void zero_f32(float* __restrict__ p, size_t n)
{
    size_t i  = (size_t)blockIdx.x * 256 + threadIdx.x;
    size_t st = (size_t)gridDim.x * 256;
    for (; i < n; i += st) p[i] = 0.f;
}

__global__ void add_kernel(const float* __restrict__ a, const float* __restrict__ b,
                           float* __restrict__ o, size_t n)
{
    size_t i  = (size_t)blockIdx.x * 256 + threadIdx.x;
    size_t st = (size_t)gridDim.x * 256;
    for (; i < n; i += st) o[i] = a[i] + b[i];
}

// ---------------------------------------------------------------------------
extern "C" void kernel_launch(void* const* d_in, const int* in_sizes, int n_in,
                              void* d_out, int out_size, void* d_ws, size_t ws_size,
                              hipStream_t stream)
{
    (void)in_sizes; (void)n_in; (void)out_size; (void)ws_size;

    const float* hidden    = (const float*)d_in[0];
    const float* ln_in_g   = (const float*)d_in[1];
    const float* ln_in_b   = (const float*)d_in[2];
    const float* Wq        = (const float*)d_in[3];
    const float* Wkv       = (const float*)d_in[4];
    const float* Wo        = (const float*)d_in[5];
    const float* ln_attn_g = (const float*)d_in[6];
    const float* ln_attn_b = (const float*)d_in[7];
    const float* ln_post_g = (const float*)d_in[8];
    const float* ln_post_b = (const float*)d_in[9];
    const float* Wg        = (const float*)d_in[10];
    const float* W1        = (const float*)d_in[11];
    const float* b1        = (const float*)d_in[12];
    const float* W2        = (const float*)d_in[13];
    const float* b2        = (const float*)d_in[14];
    float*       out       = (float*)d_out;

    // -------- workspace carve --------
    char*  ws  = (char*)d_ws;
    size_t off = 0;
    auto alloc = [&](size_t bytes) -> char* {
        char* p = ws + off;
        off += (bytes + 255) & ~(size_t)255;
        return p;
    };
    const size_t TD = (size_t)T_TOK * D_MODEL;

    float*  h_f     = (float*) alloc(TD * 4);                      // LN'd input
    __bf16* h_b     = (__bf16*)alloc(TD * 2);
    __bf16* q_b     = (__bf16*)alloc(TD * 2);                      // [T, H*DH]
    __bf16* kv_b    = (__bf16*)alloc(TD * 2 * 2);                  // [T, 2*H*DH]
    __bf16* vT      = (__bf16*)alloc((size_t)B_SZ * N_HEAD * 128 * S_LEN * 2); // padded
    float*  scores  = (float*) alloc((size_t)S_LEN * S_LEN * 4);   // per (b,h)
    __bf16* prob    = (__bf16*)alloc((size_t)S_LEN * S_LEN * 2);
    __bf16* av_b    = (__bf16*)alloc(TD * 2);                      // attn_vec
    float*  attn_o  = (float*) alloc(TD * 4);
    float*  attn_ln = (float*) alloc(TD * 4);                      // attention_output
    float*  ln_inp  = (float*) alloc(TD * 4);                      // layernorm_input
    float*  x_f     = (float*) alloc(TD * 4);
    __bf16* x_b     = (__bf16*)alloc(TD * 2);
    __bf16* Wq_t    = (__bf16*)alloc((size_t)D_MODEL * D_MODEL * 2);          // [N][K]
    __bf16* Wkv_t   = (__bf16*)alloc((size_t)2 * D_MODEL * D_MODEL * 2);
    __bf16* Wo_t    = (__bf16*)alloc((size_t)D_MODEL * D_MODEL * 2);
    __bf16* W1_t    = (__bf16*)alloc((size_t)N_EXP * DFF_SZ * D_MODEL * 2);   // [E][DFF][D]
    __bf16* W2_t    = (__bf16*)alloc((size_t)N_EXP * D_MODEL * DFF_SZ * 2);   // [E][D][DFF]
    float*  cw      = (float*) alloc((size_t)T_TOK * N_EXP * 4);
    float*  stats   = (float*) alloc(256);
    __bf16* hmid    = (__bf16*)alloc((size_t)T_TOK * DFF_SZ * 2);
    float*  moe_acc = (float*) alloc(TD * 4);

    dim3 blk(256);
    dim3 tblk(32, 8);
    auto ggrid = [](int M, int N) { return dim3((unsigned)((N + 127) / 128),
                                                (unsigned)((M + 127) / 128)); };
    auto egrid = [](size_t n) {
        size_t g = (n + 255) / 256;
        if (g > 32768) g = 32768;
        return dim3((unsigned)g);
    };

    // -------- weight cast + transpose to [N][K] bf16 --------
    cast_transpose_kernel<<<dim3(D_MODEL/32, D_MODEL/32, 1), tblk, 0, stream>>>(
        Wq, Wq_t, D_MODEL, D_MODEL, 0, 0);
    cast_transpose_kernel<<<dim3(2*D_MODEL/32, D_MODEL/32, 1), tblk, 0, stream>>>(
        Wkv, Wkv_t, D_MODEL, 2*D_MODEL, 0, 0);
    cast_transpose_kernel<<<dim3(D_MODEL/32, D_MODEL/32, 1), tblk, 0, stream>>>(
        Wo, Wo_t, D_MODEL, D_MODEL, 0, 0);
    cast_transpose_kernel<<<dim3(DFF_SZ/32, D_MODEL/32, N_EXP), tblk, 0, stream>>>(
        W1, W1_t, D_MODEL, DFF_SZ, (size_t)D_MODEL*DFF_SZ, (size_t)DFF_SZ*D_MODEL);
    cast_transpose_kernel<<<dim3(D_MODEL/32, DFF_SZ/32, N_EXP), tblk, 0, stream>>>(
        W2, W2_t, DFF_SZ, D_MODEL, (size_t)DFF_SZ*D_MODEL, (size_t)D_MODEL*DFF_SZ);

    // -------- input layernorm (eps=1e-6) --------
    layernorm_kernel<<<T_TOK, blk, 0, stream>>>(hidden, nullptr, ln_in_g, ln_in_b,
                                                1e-6f, nullptr, h_f, h_b);

    // -------- Q / KV projections --------
    gemm_wmma<EPI_BF16><<<ggrid(T_TOK, D_MODEL), blk, 0, stream>>>(
        T_TOK, D_MODEL, D_MODEL, h_b, D_MODEL, Wq_t, D_MODEL,
        nullptr, q_b, D_MODEL, 1.0f, nullptr, nullptr, 0);
    gemm_wmma<EPI_BF16><<<ggrid(T_TOK, 2 * D_MODEL), blk, 0, stream>>>(
        T_TOK, 2 * D_MODEL, D_MODEL, h_b, D_MODEL, Wkv_t, D_MODEL,
        nullptr, kv_b, 2 * D_MODEL, 1.0f, nullptr, nullptr, 0);

    // V transposed per (b,h) into padded [128][S] slices
    transpose_v_kernel<<<dim3(D_HEAD/32, S_LEN/32, B_SZ*N_HEAD), tblk, 0, stream>>>(kv_b, vT);

    // -------- attention per (b, h): scores -> softmax -> PV --------
    const float scale = 0.125f;  // 1/sqrt(64)
    for (int b = 0; b < B_SZ; b++) {
        for (int h = 0; h < N_HEAD; h++) {
            const __bf16* qp  = q_b  + (size_t)b * D_MODEL + h * D_HEAD;           // lda=B*D
            const __bf16* kp  = kv_b + (size_t)b * 2 * D_MODEL + h * 2 * D_HEAD;   // [N=S][K=DH]
            const __bf16* vtp = vT   + (size_t)(b * N_HEAD + h) * 128 * S_LEN;     // [N=DH pad][K=S]
            gemm_wmma<EPI_F32><<<ggrid(S_LEN, S_LEN), blk, 0, stream>>>(
                S_LEN, S_LEN, D_HEAD, qp, B_SZ * D_MODEL, kp, B_SZ * 2 * D_MODEL,
                scores, nullptr, S_LEN, scale, nullptr, nullptr, 0);
            softmax_row_kernel<<<S_LEN, blk, 0, stream>>>(scores, prob);
            gemm_wmma<EPI_BF16><<<ggrid(S_LEN, D_HEAD), blk, 0, stream>>>(
                S_LEN, D_HEAD, S_LEN, prob, S_LEN, vtp, S_LEN,
                nullptr, av_b + (size_t)b * D_MODEL + h * D_HEAD, B_SZ * D_MODEL,
                1.0f, nullptr, nullptr, 0);
        }
    }

    // -------- output projection --------
    gemm_wmma<EPI_F32><<<ggrid(T_TOK, D_MODEL), blk, 0, stream>>>(
        T_TOK, D_MODEL, D_MODEL, av_b, D_MODEL, Wo_t, D_MODEL,
        attn_o, nullptr, D_MODEL, 1.0f, nullptr, nullptr, 0);

    // attention_output = LN(h + attn_out, eps=1e-5)
    layernorm_kernel<<<T_TOK, blk, 0, stream>>>(h_f, attn_o, ln_attn_g, ln_attn_b,
                                                1e-5f, nullptr, attn_ln, nullptr);
    // layernorm_input = hidden + attention_output ; x = LN(., eps=1e-6)
    layernorm_kernel<<<T_TOK, blk, 0, stream>>>(hidden, attn_ln, ln_post_g, ln_post_b,
                                                1e-6f, ln_inp, x_f, x_b);

    // -------- MoE --------
    zero_f32<<<egrid(TD), blk, 0, stream>>>(moe_acc, TD);
    zero_f32<<<1, blk, 0, stream>>>(stats, 64);
    gate_kernel<<<T_TOK / 256, blk, 0, stream>>>(x_f, Wg, cw, stats);
    loss_kernel<<<1, 1, 0, stream>>>(stats, out + TD);

    for (int e = 0; e < N_EXP; e++) {
        // hmid = gelu(x @ W1[e] + b1[e]) -> bf16
        gemm_wmma<EPI_GELU_BF16><<<ggrid(T_TOK, DFF_SZ), blk, 0, stream>>>(
            T_TOK, DFF_SZ, D_MODEL, x_b, D_MODEL,
            W1_t + (size_t)e * DFF_SZ * D_MODEL, D_MODEL,
            nullptr, hmid, DFF_SZ, 1.0f, b1 + (size_t)e * DFF_SZ, nullptr, 0);
        // moe_acc += cw[:,e] * (hmid @ W2[e] + b2[e])
        gemm_wmma<EPI_ATOMIC><<<ggrid(T_TOK, D_MODEL), blk, 0, stream>>>(
            T_TOK, D_MODEL, DFF_SZ, hmid, DFF_SZ,
            W2_t + (size_t)e * D_MODEL * DFF_SZ, DFF_SZ,
            moe_acc, nullptr, D_MODEL, 1.0f, b2 + (size_t)e * D_MODEL, cw + e, N_EXP);
    }

    // output = layernorm_input + moe_out
    add_kernel<<<egrid(TD), blk, 0, stream>>>(ln_inp, moe_acc, out, TD);
}
